// TexualEmbeddingLayer_13907104104695
// MI455X (gfx1250) — matrix operand order repaired and checked by reading, not compiled, not run on hardware
//
#include <hip/hip_runtime.h>
#include <hip/hip_bf16.h>

typedef __attribute__((ext_vector_type(16))) _Float16 v16h;
typedef __attribute__((ext_vector_type(8)))  _Float16 v8h;
typedef __attribute__((ext_vector_type(8)))  float    v8f;

#define cBS 512
#define cL  128
#define cD  512
#define cK  30
#define cG  8
#define cE  2048
#define cH  1024
#define cNT (cBS*cK)    /* 15360 real tokens  */
#define cTP (cBS*32)    /* 16384 padded tokens */

__device__ __forceinline__ v8f wmma16(v16h a, v16h b, v8f c) {
  // D = A(16x32 f16) * B(32x16 f16) + C(16x16 f32)
  return __builtin_amdgcn_wmma_f32_16x16x32_f16(false, a, false, b, (short)0, c,
                                                false, false);
}
__device__ __forceinline__ v16h ldfrag(const _Float16* p) {
  return *(const v16h*)p;   // 32B per lane -> 2x global_load_b128
}

// ---------------------------------------------------------------------------
// K1: per-batch eos / length / exact top-K (value desc, index asc on ties)
// ---------------------------------------------------------------------------
__global__ void topk_kernel(const float* __restrict__ atten,
                            const int* __restrict__ text,
                            int* __restrict__ idx, int* __restrict__ lens) {
  int b = blockIdx.x * blockDim.x + threadIdx.x;
  if (b >= cBS) return;
  const int* tb = text + b * cL;
  int eos = 0, best = tb[0], cnt = 0;
  for (int j = 0; j < cL; j++) {
    int v = tb[j];
    if (v != 0) cnt++;
    if (v > best) { best = v; eos = j; }
  }
  int lb = cnt - 2; lens[b] = lb < cK ? lb : cK;
  const float* ar = atten + ((size_t)b * cL + eos) * cL;
  float pv = 3.0e38f; int pj = -1;
  for (int k = 0; k < cK; k++) {
    float bv = -3.0e38f; int bj = 0;
    for (int j = 0; j < cL; j++) {
      float m = (tb[j] != 0) ? 1.f : 0.f;
      float v = ((j == 0) || (j == eos)) ? -1.f : ar[j];
      v *= m;
      bool after = (v < pv) || (v == pv && j > pj);
      if (after && (v > bv || (v == bv && j < bj))) { bv = v; bj = j; }
    }
    idx[b * cK + k] = bj; pv = bv; pj = bj;
  }
}

// ---------------------------------------------------------------------------
// K2: gather feats rows, l2norm, emit WMMA fragments in global memory:
//     featsB : raw feats as B-frags  (B = feats^T, element (k=chan, n=token))
//     nfeatsA: l2normed feats as A-frags (A = nfeats, element (m=token,k=chan))
// 4 tokens per 256-thread block; 64 threads (8-chan chunks) per token.
// ---------------------------------------------------------------------------
__global__ void gather_pack_kernel(const float* __restrict__ features,
                                   const int* __restrict__ idx,
                                   _Float16* __restrict__ featsB,
                                   _Float16* __restrict__ nfeatsA) {
  __shared__ float ssum[4][64];
  int local = threadIdx.x >> 6, sub = threadIdx.x & 63;
  int t = blockIdx.x * 4 + local;          // token 0..15359
  int b = t / cK, k = t % cK;
  int row = idx[b * cK + k];
  int c0 = sub * 8;
  const float* src = features + ((size_t)b * cL + row) * cD + c0;
  float v[8]; float s = 0.f;
#pragma unroll
  for (int i = 0; i < 8; i++) { v[i] = src[i]; s += v[i] * v[i]; }
  ssum[local][sub] = s; __syncthreads();
  for (int off = 32; off >= 1; off >>= 1) {
    if (sub < off) ssum[local][sub] += ssum[local][sub + off];
    __syncthreads();
  }
  float rn = 1.f / (sqrtf(ssum[local][0]) + 1e-8f);
  // B-frag (raw): ntile=t/16, kstep=c0/32, lane=n%16+16*((k%16)/8), j=8*((k%32)/16)+k%8
  {
    int ntile = t >> 4, kstep = c0 >> 5;
    int lane = (t & 15) + 16 * ((c0 & 15) >> 3);
    int j = ((c0 & 31) >> 4) * 8;
    v8h o;
#pragma unroll
    for (int i = 0; i < 8; i++) o[i] = (_Float16)v[i];
    *(v8h*)(featsB + (((size_t)(ntile * 16 + kstep) * 32 + lane) * 16 + j)) = o;
  }
  // A-frag (normed): lane=m%16+16*((k%16)/8), j=8*((k%32)/16)+k%8
  {
    int mtile = t >> 4, kstep = c0 >> 5;
    int lane = (t & 15) + 16 * ((c0 & 15) >> 3);
    int j = ((c0 & 31) >> 4) * 8;
    v8h o;
#pragma unroll
    for (int i = 0; i < 8; i++) o[i] = (_Float16)(v[i] * rn);
    *(v8h*)(nfeatsA + (((size_t)(mtile * 16 + kstep) * 32 + lane) * 16 + j)) = o;
  }
}

// ---------------------------------------------------------------------------
// K3: generic weight pack -> B fragments.  src[k][n] f32, K padded with 0.
// ---------------------------------------------------------------------------
__global__ void packB_kernel(const float* __restrict__ src,
                             _Float16* __restrict__ dst,
                             int Ksrc, int Kpad, int N) {
  int tid = blockIdx.x * blockDim.x + threadIdx.x;
  int kpb = Kpad >> 3;
  if (tid >= N * kpb) return;
  int kb = tid % kpb, n = tid / kpb;
  int k0 = kb * 8;
  int ntile = n >> 4, kstep = k0 >> 5, ksteps = Kpad >> 5;
  int lane = (n & 15) + 16 * ((k0 & 15) >> 3);
  int j = ((k0 & 31) >> 4) * 8;
  v8h o;
#pragma unroll
  for (int i = 0; i < 8; i++) {
    int k = k0 + i;
    o[i] = (_Float16)((k < Ksrc) ? src[(size_t)k * N + n] : 0.f);
  }
  *(v8h*)(dst + (((size_t)(ntile * ksteps + kstep) * 32 + lane) * 16 + j)) = o;
}

// K3b: dlp_w [8,512] -> A fragments, M padded 8->16 with zeros.
__global__ void pack_dlpw_kernel(const float* __restrict__ w,
                                 _Float16* __restrict__ dst) {
  int tid = blockIdx.x * blockDim.x + threadIdx.x;   // 16*64
  if (tid >= 16 * 64) return;
  int m = tid >> 6, kb = tid & 63; int k0 = kb * 8;
  int kstep = k0 >> 5;
  int lane = (m & 15) + 16 * ((k0 & 15) >> 3);
  int j = ((k0 & 31) >> 4) * 8;
  v8h o;
#pragma unroll
  for (int i = 0; i < 8; i++)
    o[i] = (_Float16)((m < cG) ? w[m * cD + k0 + i] : 0.f);
  *(v8h*)(dst + (((size_t)kstep * 32 + lane) * 16 + j)) = o;
}

// ---------------------------------------------------------------------------
// K4: group GEMM1: h[8,15360] = dlp_w[8pad16,512] @ feats^T  (+ dlp_b)
// 8 waves * 2 n-tiles per block; A reloaded per k-step (L2-hot) -> no spills.
// ---------------------------------------------------------------------------
__global__ void ggemm1_kernel(const _Float16* __restrict__ A,
                              const _Float16* __restrict__ B,
                              const float* __restrict__ dlp_b,
                              _Float16* __restrict__ h_plain) {
  int lane = threadIdx.x & 31, wv = threadIdx.x >> 5;
  int nb = blockIdx.x * 16 + wv * 2;       // two n-tiles: nb, nb+1
  v8f acc0 = {}, acc1 = {};
#pragma unroll 1
  for (int ks = 0; ks < 16; ks++) {
    v16h a  = ldfrag(A + ((size_t)ks * 32 + lane) * 16);
    v16h b0 = ldfrag(B + (((size_t)(nb * 16 + ks) * 32 + lane) * 16));
    v16h b1 = ldfrag(B + (((size_t)((nb + 1) * 16 + ks) * 32 + lane) * 16));
    acc0 = wmma16(a, b0, acc0);
    acc1 = wmma16(a, b1, acc1);
  }
  if (lane < 16) {                         // rows m=0..7 valid (A rows 8..15 zero)
#pragma unroll
    for (int i = 0; i < 8; i++) {
      float bv = dlp_b[i];
      h_plain[(size_t)i * cNT + nb * 16 + lane]       = (_Float16)(acc0[i] + bv);
      h_plain[(size_t)i * cNT + (nb + 1) * 16 + lane] = (_Float16)(acc1[i] + bv);
    }
  }
}

// K5: repack h into A-frags Hp[512, 240pad256]: Hp[b][t] = h[b%8][(b/8)*240+t]
__global__ void pack_hp_kernel(const _Float16* __restrict__ h_plain,
                               _Float16* __restrict__ dst) {
  int tid = blockIdx.x * blockDim.x + threadIdx.x;   // 512*32
  if (tid >= 512 * 32) return;
  int bi = tid >> 5, kb = tid & 31; int k0 = kb * 8;
  int mtile = bi >> 4, kstep = k0 >> 5;
  int lane = (bi & 15) + 16 * ((k0 & 15) >> 3);
  int j = ((k0 & 31) >> 4) * 8;
  const _Float16* hs = h_plain + (size_t)(bi & 7) * cNT + (bi >> 3) * (cG * cK);
  v8h o;
#pragma unroll
  for (int i = 0; i < 8; i++) {
    int k = k0 + i;
    o[i] = (k < cG * cK) ? hs[k] : (_Float16)0.f;
  }
  *(v8h*)(dst + (((size_t)(mtile * 8 + kstep) * 32 + lane) * 16 + j)) = o;
}

// K6: group GEMM2: Y[512,2048] = Hp[512,256] @ dlp_lw (+dlp_lb)
// grid (32 mtiles, 8 nchunks); 8 waves * 2 n-tiles.
__global__ void ggemm2_kernel(const _Float16* __restrict__ A,
                              const _Float16* __restrict__ B,
                              const float* __restrict__ bias,
                              float* __restrict__ Y) {
  int lane = threadIdx.x & 31, wv = threadIdx.x >> 5;
  int mtile = blockIdx.x;                    // < 32
  int nb = blockIdx.y * 16 + wv * 2;         // two n-tiles
  v8f acc0 = {}, acc1 = {};
#pragma unroll 1
  for (int ks = 0; ks < 8; ks++) {
    v16h a  = ldfrag(A + (((size_t)(mtile * 8 + ks) * 32 + lane) * 16));
    v16h b0 = ldfrag(B + (((size_t)(nb * 8 + ks) * 32 + lane) * 16));
    v16h b1 = ldfrag(B + (((size_t)((nb + 1) * 8 + ks) * 32 + lane) * 16));
    acc0 = wmma16(a, b0, acc0);
    acc1 = wmma16(a, b1, acc1);
  }
  int mrow = mtile * 16 + 8 * (lane >> 4);
  int n0 = nb * 16 + (lane & 15);
  int n1 = (nb + 1) * 16 + (lane & 15);
  float bv0 = bias[n0], bv1 = bias[n1];
#pragma unroll
  for (int i = 0; i < 8; i++) {
    Y[(size_t)(mrow + i) * cE + n0] = acc0[i] + bv0;
    Y[(size_t)(mrow + i) * cE + n1] = acc1[i] + bv1;
  }
}

// K7: per-row l2-norm factor of Y
__global__ void rownorm_kernel(const float* __restrict__ Y, float* __restrict__ rn) {
  __shared__ float s[256];
  int b = blockIdx.x;
  float acc = 0.f;
  for (int i = threadIdx.x; i < cE; i += 256) {
    float v = Y[(size_t)b * cE + i]; acc += v * v;
  }
  s[threadIdx.x] = acc; __syncthreads();
  for (int off = 128; off >= 1; off >>= 1) {
    if ((int)threadIdx.x < off) s[threadIdx.x] += s[threadIdx.x + off];
    __syncthreads();
  }
  if (threadIdx.x == 0) rn[b] = 1.f / (sqrtf(s[0]) + 1e-8f);
}

// K8: MLP GEMM1: x1[15360,1024] = nfeats @ w1 (+b1), f16 plain output
// 8 waves; wave holds 8 acc tiles (64 VGPRs); A double-buffered.
__global__ void mgemm1_kernel(const _Float16* __restrict__ A,
                              const _Float16* __restrict__ B,
                              const float* __restrict__ b1,
                              _Float16* __restrict__ x1) {
  int lane = threadIdx.x & 31, wv = threadIdx.x >> 5;
  int mtile = blockIdx.x;                    // < 960
  const _Float16* pA = A + (size_t)mtile * 16 * 512 + lane * 16;   // +512/kstep
  v8f acc[8] = {};
  v16h a = ldfrag(pA);
#pragma unroll 1
  for (int ks = 0; ks < 16; ks++) {
    v16h an;
    if (ks + 1 < 16) an = ldfrag(pA + (size_t)(ks + 1) * 512);
#pragma unroll
    for (int q = 0; q < 8; q++) {
      int ntile = wv * 8 + q;                // < 64
      v16h bb = ldfrag(B + (((size_t)(ntile * 16 + ks) * 32 + lane) * 16));
      acc[q] = wmma16(a, bb, acc[q]);
    }
    a = an;
  }
  for (int q = 0; q < 8; q++) {
    int n = (wv * 8 + q) * 16 + (lane & 15);
    int mrow = mtile * 16 + 8 * (lane >> 4);
    float bv = b1[n];
#pragma unroll
    for (int i = 0; i < 8; i++)
      x1[(size_t)(mrow + i) * cH + n] = (_Float16)(acc[q][i] + bv);
  }
}

// K9: BatchNorm train-stats per channel (deterministic tree reduce)
__global__ void bnstats_kernel(const _Float16* __restrict__ x1,
                               const float* __restrict__ bn_g,
                               const float* __restrict__ bn_b,
                               float* __restrict__ scale,
                               float* __restrict__ shift) {
  __shared__ float s1[256], s2[256];
  int ch = blockIdx.x;
  float a = 0.f, b = 0.f;
  for (int t = threadIdx.x; t < cNT; t += 256) {
    float v = (float)x1[(size_t)t * cH + ch];
    a += v; b += v * v;
  }
  s1[threadIdx.x] = a; s2[threadIdx.x] = b; __syncthreads();
  for (int off = 128; off >= 1; off >>= 1) {
    if ((int)threadIdx.x < off) {
      s1[threadIdx.x] += s1[threadIdx.x + off];
      s2[threadIdx.x] += s2[threadIdx.x + off];
    }
    __syncthreads();
  }
  if (threadIdx.x == 0) {
    float mu = s1[0] / (float)cNT;
    float var = s2[0] / (float)cNT - mu * mu;
    float sc = bn_g[ch] * rsqrtf(var + 1e-5f);
    scale[ch] = sc; shift[ch] = bn_b[ch] - mu * sc;
  }
}

// K10: BN+ReLU fused into A-fragment repack, token dim padded 30->32/batch
__global__ void pack_xn_kernel(const _Float16* __restrict__ x1,
                               const float* __restrict__ scale,
                               const float* __restrict__ shift,
                               _Float16* __restrict__ dst) {
  int tid = blockIdx.x * blockDim.x + threadIdx.x;   // cTP * 128
  if (tid >= cTP * (cH / 8)) return;
  int tp = tid >> 7, kb = tid & 127; int k0 = kb * 8;
  int b = tp >> 5, kk = tp & 31;
  v8h o = {};
  if (kk < cK) {
    const _Float16* x = x1 + (size_t)(b * cK + kk) * cH + k0;
#pragma unroll
    for (int i = 0; i < 8; i++) {
      float xv = (float)x[i] * scale[k0 + i] + shift[k0 + i];
      o[i] = (_Float16)fmaxf(xv, 0.f);
    }
  }
  int mtile = tp >> 4, kstep = k0 >> 5;
  int lane = (tp & 15) + 16 * ((k0 & 15) >> 3);
  int j = ((k0 & 31) >> 4) * 8;
  *(v8h*)(dst + (((size_t)(mtile * 32 + kstep) * 32 + lane) * 16 + j)) = o;
}

// K11: MLP GEMM2 + fused masked max-pool over K rows + add l2norm(new_base)
// grid (batch, nchunk of 64 cols); 8 waves: mt = w&1, ntile = nchunk*4+(w>>1)
// Double-buffered A/B fragment stream: loads for ks+1 issue before wmma(ks).
__global__ void mgemm2_pool_kernel(const _Float16* __restrict__ A,
                                   const _Float16* __restrict__ B,
                                   const float* __restrict__ b2,
                                   const int* __restrict__ lens,
                                   const float* __restrict__ Y,
                                   const float* __restrict__ rownorm,
                                   float* __restrict__ out) {
  __shared__ float smax[4][16];
  int lane = threadIdx.x & 31, wv = threadIdx.x >> 5;
  int b = blockIdx.x, nchunk = blockIdx.y;
  int mt = wv & 1, ntl = wv >> 1;
  int ntile = nchunk * 4 + ntl;              // < 128
  int mtileP = b * 2 + mt;                   // < 1024
  const _Float16* pA = A + (size_t)mtileP * 32 * 512 + lane * 16;  // +512/kstep
  const _Float16* pB = B + (size_t)ntile * 32 * 512 + lane * 16;
  v8f acc = {};
  v16h a = ldfrag(pA);
  v16h bb = ldfrag(pB);
#pragma unroll 1
  for (int ks = 0; ks < 31; ks++) {
    v16h an = ldfrag(pA + (size_t)(ks + 1) * 512);
    v16h bn = ldfrag(pB + (size_t)(ks + 1) * 512);
    __builtin_prefetch(pB + (size_t)(ks + 2) * 512, 0, 1);
    acc = wmma16(a, bb, acc);
    a = an; bb = bn;
  }
  acc = wmma16(a, bb, acc);
  int n = ntile * 16 + (lane & 15);
  float bv = b2[n];
  int lb = lens[b];
  int mbase = mt * 16 + 8 * (lane >> 4);
  float vmax = -3.0e38f;
#pragma unroll
  for (int i = 0; i < 8; i++) {
    float val = acc[i] + bv;
    if (mbase + i < lb) vmax = fmaxf(vmax, val);
  }
  vmax = fmaxf(vmax, __shfl_xor(vmax, 16, 32));   // merge lane halves (same n)
  if (mt == 0) { if (lane < 16) smax[ntl][lane] = vmax; }
  __syncthreads();
  if (mt == 1 && lane < 16) {
    float pooled = fmaxf(vmax, smax[ntl][lane]);
    out[(size_t)b * cE + n] = pooled + Y[(size_t)b * cE + n] * rownorm[b];
  }
}

// ---------------------------------------------------------------------------
extern "C" void kernel_launch(void* const* d_in, const int* in_sizes, int n_in,
                              void* d_out, int out_size, void* d_ws, size_t ws_size,
                              hipStream_t stream) {
  (void)in_sizes; (void)n_in; (void)out_size;
  const float* features = (const float*)d_in[0];
  const float* atten    = (const float*)d_in[1];
  const int*   text     = (const int*)d_in[2];
  const float* dlp_w    = (const float*)d_in[4];
  const float* dlp_b    = (const float*)d_in[5];
  const float* dlp_lw   = (const float*)d_in[6];
  const float* dlp_lb   = (const float*)d_in[7];
  const float* mlp_w1   = (const float*)d_in[8];
  const float* mlp_b1   = (const float*)d_in[9];
  const float* bn_g     = (const float*)d_in[10];
  const float* bn_b     = (const float*)d_in[11];
  const float* mlp_w2   = (const float*)d_in[12];
  const float* mlp_b2   = (const float*)d_in[13];
  float* out = (float*)d_out;

  char* ws = (char*)d_ws;
  size_t off = 0;
  auto take = [&](size_t bytes) -> char* {
    char* p = ws + off;
    off += (bytes + 255) & ~(size_t)255;
    return p;
  };
  int*       idx      = (int*)      take((size_t)cNT * 4);
  int*       lens     = (int*)      take((size_t)cBS * 4);
  _Float16*  featsB   = (_Float16*) take((size_t)cNT * cD * 2);        // B-frags
  _Float16*  nfeatsA  = (_Float16*) take((size_t)cNT * cD * 2);        // A-frags
  _Float16*  dlpwA    = (_Float16*) take((size_t)16 * cD * 2);         // 16x512
  _Float16*  dlplwB   = (_Float16*) take((size_t)256 * cE * 2);        // 256x2048
  _Float16*  w1B      = (_Float16*) take((size_t)cD * cH * 2);
  _Float16*  w2B      = (_Float16*) take((size_t)cH * cE * 2);
  _Float16*  h_plain  = (_Float16*) take((size_t)cG * cNT * 2);
  _Float16*  hpA      = (_Float16*) take((size_t)512 * 256 * 2);
  float*     Yb       = (float*)    take((size_t)cBS * cE * 4);
  float*     rnorm    = (float*)    take((size_t)cBS * 4);
  _Float16*  x1       = (_Float16*) take((size_t)cNT * cH * 2);
  float*     bnscale  = (float*)    take((size_t)cH * 4);
  float*     bnshift  = (float*)    take((size_t)cH * 4);
  _Float16*  xnA      = (_Float16*) take((size_t)cTP * cH * 2);
  if (off > ws_size) return;

  topk_kernel<<<2, 256, 0, stream>>>(atten, text, idx, lens);
  gather_pack_kernel<<<cNT / 4, 256, 0, stream>>>(features, idx, featsB, nfeatsA);

  packB_kernel<<<(cE * (256 / 8) + 255) / 256, 256, 0, stream>>>(dlp_lw, dlplwB, 240, 256, cE);
  packB_kernel<<<(cH * (cD / 8) + 255) / 256, 256, 0, stream>>>(mlp_w1, w1B, cD, cD, cH);
  packB_kernel<<<(cE * (cH / 8) + 255) / 256, 256, 0, stream>>>(mlp_w2, w2B, cH, cH, cE);
  pack_dlpw_kernel<<<4, 256, 0, stream>>>(dlp_w, dlpwA);

  ggemm1_kernel<<<60, 256, 0, stream>>>(dlpwA, featsB, dlp_b, h_plain);
  pack_hp_kernel<<<64, 256, 0, stream>>>(h_plain, hpA);
  ggemm2_kernel<<<dim3(32, 8), 256, 0, stream>>>(hpA, dlplwB, dlp_lb, Yb);
  rownorm_kernel<<<cBS, 256, 0, stream>>>(Yb, rnorm);

  mgemm1_kernel<<<cNT / 16, 256, 0, stream>>>(nfeatsA, w1B, mlp_b1, x1);
  bnstats_kernel<<<cH, 256, 0, stream>>>(x1, bn_g, bn_b, bnscale, bnshift);
  pack_xn_kernel<<<(cTP * (cH / 8)) / 256, 256, 0, stream>>>(x1, bnscale, bnshift, xnA);

  mgemm2_pool_kernel<<<dim3(cBS, cE / 64), 256, 0, stream>>>(
      xnA, w2B, mlp_b2, lens, Yb, rnorm, out);
}
